// MultiHeadAttention_77446850281793
// MI455X (gfx1250) — matrix operand order repaired and checked by reading, not compiled, not run on hardware
//
#include <hip/hip_runtime.h>

#define SEQ   2048
#define DIM   1024
#define HEADS 16
#define DK    64

typedef __attribute__((ext_vector_type(16))) __bf16         v16bf;
typedef __attribute__((ext_vector_type(8)))  float          v8f;
typedef __attribute__((ext_vector_type(8)))  unsigned short u16x8;
typedef __attribute__((ext_vector_type(4)))  float          f32x4;
typedef __attribute__((ext_vector_type(4)))  unsigned int   u32x4;
typedef __attribute__((ext_vector_type(8)))  int            i32x8;
typedef __attribute__((ext_vector_type(4)))  int            i32x4;

union FragBF { v16bf v; unsigned short u[16]; u16x8 q[2]; };
union FragF  { v8f   v; float f[8]; };

__device__ __forceinline__ unsigned short f2bf(float f) {
  union { float f; unsigned u; } x; x.f = f;
  unsigned u = x.u;
  u += 0x7FFFu + ((u >> 16) & 1u);          // round-to-nearest-even
  return (unsigned short)(u >> 16);
}

__device__ __forceinline__ v8f wmma_bf16(v16bf a, v16bf b, v8f c) {
  // D = A(16x32 bf16) x B(32x16 bf16) + C(16x16 f32)
  return __builtin_amdgcn_wmma_f32_16x16x32_bf16(false, a, false, b,
                                                 (short)0, c, false, false);
}

// ---------------------------------------------------------------------------
// One-shot fp32 -> bf16 conversion (memory-bound; n multiple of 8).
// ---------------------------------------------------------------------------
__global__ __launch_bounds__(256) void cvt_bf16(const float* __restrict__ src,
                                                unsigned short* __restrict__ dst,
                                                int n) {
  int idx = (blockIdx.x * 256 + threadIdx.x) * 8;
  if (idx >= n) return;
  f32x4 a = *(const f32x4*)(src + idx);
  f32x4 b = *(const f32x4*)(src + idx + 4);
  u16x8 o;
  o[0] = f2bf(a[0]); o[1] = f2bf(a[1]); o[2] = f2bf(a[2]); o[3] = f2bf(a[3]);
  o[4] = f2bf(b[0]); o[5] = f2bf(b[1]); o[6] = f2bf(b[2]); o[7] = f2bf(b[3]);
  *(u16x8*)(dst + idx) = o;
}

// ---------------------------------------------------------------------------
// Build TDM descriptor group1 for a 2-D bf16 tile:
//   tile (tileD0 x tileD1) from a row-major tensor with row stride strideElems.
// Bit layout per CDNA5 ISA 8.4 (data_size=1 -> 2 bytes).
// ---------------------------------------------------------------------------
__device__ __forceinline__ i32x8 tdm_group1(int tensorD0, int tensorD1,
                                            int tileD0, int tileD1,
                                            int strideElems) {
  i32x8 g;
  g[0] = (int)(1u << 16);                                     // data_size = 2B
  g[1] = (int)((unsigned)(tensorD0 & 0xFFFF) << 16);          // tdim0[15:0] @63:48
  g[2] = (int)(((unsigned)tensorD0 >> 16) & 0xFFFFu) |
         (int)((unsigned)(tensorD1 & 0xFFFF) << 16);          // tdim0 hi | tdim1 lo
  g[3] = (int)(((unsigned)tensorD1 >> 16) & 0xFFFFu) |
         (int)((unsigned)(tileD0 & 0xFFFF) << 16);            // tdim1 hi | tile_dim0
  g[4] = (int)(tileD1 & 0xFFFF);                              // tile_dim1 (tile_dim2=0)
  g[5] = strideElems;                                         // tensor_dim0_stride lo32
  g[6] = 0;                                                   // stride hi16 | s1 lo16
  g[7] = 0;
  return g;
}

__device__ __forceinline__ u32x4 tdm_group0(unsigned ldsAddr,
                                            unsigned long long gAddr) {
  u32x4 g;
  g[0] = 1u;                                                  // count=1, user mode
  g[1] = ldsAddr;                                             // lds_addr (bytes)
  g[2] = (unsigned)gAddr;                                     // global_addr[31:0]
  g[3] = ((unsigned)(gAddr >> 32) & 0x01FFFFFFu) | (2u << 30);// addr[56:32] | type=2
  return g;
}

// ---------------------------------------------------------------------------
// C(MxN) = X(MxK) @ W(NxK)^T + bias(N).  X,W bf16 (raw u16), bias fp32.
// Block: 256 threads (8 waves, 4x2 wave grid), tile 128x64, K step 64.
// Tiles staged global->LDS by the Tensor Data Mover (wave 0 issues, TENSORcnt).
// ---------------------------------------------------------------------------
template <typename OT>
__global__ __launch_bounds__(256) void gemm_xwt_bf16(
    const unsigned short* __restrict__ X, const unsigned short* __restrict__ W,
    const float* __restrict__ bias, OT* __restrict__ C, int M, int N, int K) {
  constexpr int BM = 128, BN = 64, BK = 64;
  __shared__ unsigned short Xs[BM * BK];   // 16 KB, row-major [m][k]
  __shared__ unsigned short Ws[BN * BK];   //  8 KB, row-major [n][k]

  const int t    = threadIdx.x;
  const int m0   = blockIdx.y * BM;
  const int n0   = blockIdx.x * BN;
  const int w    = t >> 5, lane = t & 31;
  const int wm   = w >> 1, wn = w & 1;         // 4x2 waves -> 32x32 per wave
  const int lm   = lane & 15, hiHalf = lane >> 4;

  const i32x8 g1x = tdm_group1(K, BM, BK, BM, K);
  const i32x8 g1w = tdm_group1(K, BN, BK, BN, K);
  const i32x4 gz4 = {0, 0, 0, 0};
  const i32x8 gz8 = {0, 0, 0, 0, 0, 0, 0, 0};
  const unsigned ldsXs = (unsigned)(size_t)&Xs[0];
  const unsigned ldsWs = (unsigned)(size_t)&Ws[0];
  const unsigned long long xBase =
      (unsigned long long)X + ((size_t)m0 * K) * 2ull;
  const unsigned long long wBase =
      (unsigned long long)W + ((size_t)n0 * K) * 2ull;

  FragF acc[2][2];
#pragma unroll
  for (int i = 0; i < 2; ++i)
#pragma unroll
    for (int j = 0; j < 2; ++j)
#pragma unroll
      for (int r = 0; r < 8; ++r) acc[i][j].f[r] = 0.0f;

#pragma unroll 1
  for (int kk = 0; kk < K; kk += BK) {
    if (w == 0) {                       // one wave drives the DMA engine
      __builtin_amdgcn_tensor_load_to_lds(
          tdm_group0(ldsXs, xBase + (unsigned long long)kk * 2ull), g1x,
          gz4, gz4, gz8, 0);
      __builtin_amdgcn_tensor_load_to_lds(
          tdm_group0(ldsWs, wBase + (unsigned long long)kk * 2ull), g1w,
          gz4, gz4, gz8, 0);
      __builtin_amdgcn_s_wait_tensorcnt(0);
    }
    __syncthreads();

    FragBF a[2][2], b[2][2];
#pragma unroll
    for (int qm = 0; qm < 2; ++qm)      // A frag: lane=row, K runs {0..7,16..23}+koff
#pragma unroll
      for (int dc = 0; dc < 2; ++dc) {
        int base = (wm * 32 + qm * 16 + lm) * BK + dc * 32 + hiHalf * 8;
        a[qm][dc].q[0] = *(const u16x8*)&Xs[base];
        a[qm][dc].q[1] = *(const u16x8*)&Xs[base + 16];
      }
#pragma unroll
    for (int nn = 0; nn < 2; ++nn)      // B frag: lane=col n, K runs {0..7,16..23}+koff
#pragma unroll
      for (int dc = 0; dc < 2; ++dc) {
        int base = (wn * 32 + nn * 16 + lm) * BK + dc * 32 + hiHalf * 8;
        b[nn][dc].q[0] = *(const u16x8*)&Ws[base];
        b[nn][dc].q[1] = *(const u16x8*)&Ws[base + 16];
      }
#pragma unroll
    for (int qm = 0; qm < 2; ++qm)
#pragma unroll
      for (int nn = 0; nn < 2; ++nn)
#pragma unroll
        for (int dc = 0; dc < 2; ++dc)
          acc[qm][nn].v = wmma_bf16(a[qm][dc].v, b[nn][dc].v, acc[qm][nn].v);

    __syncthreads();
  }

#pragma unroll
  for (int qm = 0; qm < 2; ++qm)
#pragma unroll
    for (int nn = 0; nn < 2; ++nn) {
      int   ncol = n0 + wn * 32 + nn * 16 + lm;
      float bv   = bias[ncol];
#pragma unroll
      for (int r = 0; r < 8; ++r) {
        int   m = m0 + wm * 32 + qm * 16 + r + hiHalf * 8;  // C/D layout
        float v = acc[qm][nn].f[r] + bv;
        if constexpr (sizeof(OT) == 4) C[(size_t)m * N + ncol] = v;
        else                           C[(size_t)m * N + ncol] = f2bf(v);
      }
    }
}

// ---------------------------------------------------------------------------
// Attention with softmax across the HEADS axis (reference quirk).
// Block: 512 threads = 16 waves, wave h owns head h. q-tile 32, k-tile 32.
// Dynamic LDS: denom 32x32 f32 (cross-wave) + per-wave attn slab + per-wave V^T.
// ---------------------------------------------------------------------------
__global__ __launch_bounds__(512) void attn_head_softmax(
    const unsigned short* __restrict__ Qp, const unsigned short* __restrict__ Kp,
    const unsigned short* __restrict__ Vp, unsigned short* __restrict__ ctx) {
  extern __shared__ char smem_raw[];
  float*          denom = (float*)smem_raw;                                // 4 KB
  unsigned short* attnS = (unsigned short*)(smem_raw + 4096);              // 32 KB
  unsigned short* vtAll = (unsigned short*)(smem_raw + 4096 + 32768);      // 64 KB

  const int tid  = threadIdx.x;
  const int h    = tid >> 5, lane = tid & 31;
  const int lm   = lane & 15, hiHalf = lane >> 4;
  const int q0   = blockIdx.x * 32;
  const int hc   = h * DK;
  unsigned short* vts = vtAll + h * (DK * 32);     // wave-private V^T [d][k]
  unsigned short* as  = attnS + h * (32 * 32);     // wave-private attn [q][k]

  // Loop-invariant Q fragments (A layout matches row-major global directly).
  FragBF qa[2][2];
#pragma unroll
  for (int qm = 0; qm < 2; ++qm)
#pragma unroll
    for (int dc = 0; dc < 2; ++dc) {
      const unsigned short* p =
          Qp + (size_t)(q0 + qm * 16 + lm) * DIM + hc + dc * 32 + hiHalf * 8;
      qa[qm][dc].q[0] = *(const u16x8*)p;
      qa[qm][dc].q[1] = *(const u16x8*)(p + 16);
    }

  FragF cacc[2][4];
#pragma unroll
  for (int i = 0; i < 2; ++i)
#pragma unroll
    for (int j = 0; j < 4; ++j)
#pragma unroll
      for (int r = 0; r < 8; ++r) cacc[i][j].f[r] = 0.0f;

#pragma unroll 1
  for (int kt = 0; kt < SEQ; kt += 32) {
    // K fragments (B layout matches row-major global directly).
    FragBF kb[2][2];
#pragma unroll
    for (int kn = 0; kn < 2; ++kn)
#pragma unroll
      for (int dc = 0; dc < 2; ++dc) {
        const unsigned short* p =
            Kp + (size_t)(kt + kn * 16 + lm) * DIM + hc + dc * 32 + hiHalf * 8;
        kb[kn][dc].q[0] = *(const u16x8*)p;
        kb[kn][dc].q[1] = *(const u16x8*)(p + 16);
      }

    // scores S = Q @ K^T for this head's 32x32 tile
    FragF sacc[2][2];
#pragma unroll
    for (int qm = 0; qm < 2; ++qm)
#pragma unroll
      for (int kn = 0; kn < 2; ++kn) {
#pragma unroll
        for (int r = 0; r < 8; ++r) sacc[qm][kn].f[r] = 0.0f;
#pragma unroll
        for (int dc = 0; dc < 2; ++dc)
          sacc[qm][kn].v = wmma_bf16(qa[qm][dc].v, kb[kn][dc].v, sacc[qm][kn].v);
      }

    // Stage V^T into wave-private LDS (lane owns seq row kt+lane).
    union { u16x8 q[8]; unsigned short s[64]; } vld;
    const u16x8* vq = (const u16x8*)(Vp + (size_t)(kt + lane) * DIM + hc);
#pragma unroll
    for (int i = 0; i < 8; ++i) vld.q[i] = vq[i];
#pragma unroll
    for (int d = 0; d < 64; ++d) vts[d * 32 + lane] = vld.s[d];

    if (kt + 32 < SEQ) {  // global_prefetch_b8 of next tile
      __builtin_prefetch(Kp + (size_t)(kt + 32 + lane) * DIM + hc, 0, 0);
      __builtin_prefetch(Vp + (size_t)(kt + 32 + lane) * DIM + hc, 0, 0);
    }

    // Cross-head softmax denominator (the only cross-wave state).
    __syncthreads();                       // prior iter's denom reads done
    denom[tid] = 0.0f; denom[tid + 512] = 0.0f;
    __syncthreads();

    float ex[2][2][8];
#pragma unroll
    for (int qm = 0; qm < 2; ++qm)
#pragma unroll
      for (int kn = 0; kn < 2; ++kn)
#pragma unroll
        for (int r = 0; r < 8; ++r) {
          float e = __expf(sacc[qm][kn].f[r] * 0.125f);   // scale = 1/sqrt(64)
          ex[qm][kn][r] = e;
          atomicAdd(&denom[(qm * 16 + r + hiHalf * 8) * 32 + kn * 16 + lm], e);
        }
    __syncthreads();

    // Normalize, stash attn (bf16) in wave-private slab to re-layout as A frags.
#pragma unroll
    for (int qm = 0; qm < 2; ++qm)
#pragma unroll
      for (int kn = 0; kn < 2; ++kn)
#pragma unroll
        for (int r = 0; r < 8; ++r) {
          int qq = qm * 16 + r + hiHalf * 8, kx = kn * 16 + lm;
          float a = ex[qm][kn][r] * __builtin_amdgcn_rcpf(denom[qq * 32 + kx]);
          as[qq * 32 + kx] = f2bf(a);
        }

    FragBF aa[2];   // same-wave DS ordering guarantees visibility, no barrier
#pragma unroll
    for (int qm = 0; qm < 2; ++qm) {
      int base = (qm * 16 + lm) * 32 + hiHalf * 8;
      aa[qm].q[0] = *(const u16x8*)&as[base];
      aa[qm].q[1] = *(const u16x8*)&as[base + 16];
    }

    // ctx += attn @ V   (B frags from V^T slab, K runs {0..7,16..23})
#pragma unroll
    for (int dn = 0; dn < 4; ++dn) {
      FragBF vb;
      int base = (dn * 16 + lm) * 32 + hiHalf * 8;
      vb.q[0] = *(const u16x8*)&vts[base];
      vb.q[1] = *(const u16x8*)&vts[base + 16];
#pragma unroll
      for (int qm = 0; qm < 2; ++qm)
        cacc[qm][dn].v = wmma_bf16(aa[qm].v, vb.v, cacc[qm][dn].v);
    }
  }

  // Write ctx (concat layout: [s][h*64+d]) as bf16.
#pragma unroll
  for (int qm = 0; qm < 2; ++qm)
#pragma unroll
    for (int dn = 0; dn < 4; ++dn)
#pragma unroll
      for (int r = 0; r < 8; ++r) {
        int m = q0 + qm * 16 + r + hiHalf * 8;
        int d = dn * 16 + lm;
        ctx[(size_t)m * DIM + hc + d] = f2bf(cacc[qm][dn].f[r]);
      }
}

// ---------------------------------------------------------------------------
extern "C" void kernel_launch(void* const* d_in, const int* in_sizes, int n_in,
                              void* d_out, int out_size, void* d_ws, size_t ws_size,
                              hipStream_t stream) {
  (void)in_sizes; (void)n_in; (void)out_size; (void)ws_size;
  const float* q  = (const float*)d_in[0];
  const float* k  = (const float*)d_in[1];
  const float* v  = (const float*)d_in[2];
  const float* Wq = (const float*)d_in[3];
  const float* bq = (const float*)d_in[4];
  const float* Wk = (const float*)d_in[5];
  const float* bk = (const float*)d_in[6];
  const float* Wv = (const float*)d_in[7];
  const float* bv = (const float*)d_in[8];
  const float* Wo = (const float*)d_in[9];
  const float* bo = (const float*)d_in[10];

  const size_t IN  = (size_t)SEQ * DIM;       // 2M elems
  const size_t WSZ = (size_t)DIM * DIM;       // 1M elems
  unsigned short* p   = (unsigned short*)d_ws;
  unsigned short* qb  = p;              p += IN;
  unsigned short* kb2 = p;              p += IN;
  unsigned short* vb2 = p;              p += IN;
  unsigned short* wqb = p;              p += WSZ;
  unsigned short* wkb = p;              p += WSZ;
  unsigned short* wvb = p;              p += WSZ;
  unsigned short* wob = p;              p += WSZ;
  unsigned short* Qp  = p;              p += IN;
  unsigned short* Kp  = p;              p += IN;
  unsigned short* Vp  = p;              p += IN;
  unsigned short* ctx = p;              // total 36 MB of ws

  // One-shot fp32 -> bf16 conversions (memory-bound).
  dim3 cblk(256);
  cvt_bf16<<<dim3((unsigned)(IN / 8 / 256)),  cblk, 0, stream>>>(q,  qb,  (int)IN);
  cvt_bf16<<<dim3((unsigned)(IN / 8 / 256)),  cblk, 0, stream>>>(k,  kb2, (int)IN);
  cvt_bf16<<<dim3((unsigned)(IN / 8 / 256)),  cblk, 0, stream>>>(v,  vb2, (int)IN);
  cvt_bf16<<<dim3((unsigned)(WSZ / 8 / 256)), cblk, 0, stream>>>(Wq, wqb, (int)WSZ);
  cvt_bf16<<<dim3((unsigned)(WSZ / 8 / 256)), cblk, 0, stream>>>(Wk, wkb, (int)WSZ);
  cvt_bf16<<<dim3((unsigned)(WSZ / 8 / 256)), cblk, 0, stream>>>(Wv, wvb, (int)WSZ);
  cvt_bf16<<<dim3((unsigned)(WSZ / 8 / 256)), cblk, 0, stream>>>(Wo, wob, (int)WSZ);

  dim3 blk(256);
  dim3 grd(DIM / 64, SEQ / 128);              // (16,16)
  gemm_xwt_bf16<unsigned short><<<grd, blk, 0, stream>>>(qb,  wqb, bq, Qp, SEQ, DIM, DIM);
  gemm_xwt_bf16<unsigned short><<<grd, blk, 0, stream>>>(kb2, wkb, bk, Kp, SEQ, DIM, DIM);
  gemm_xwt_bf16<unsigned short><<<grd, blk, 0, stream>>>(vb2, wvb, bv, Vp, SEQ, DIM, DIM);

  size_t smem = 4096 + 16 * 32 * 32 * 2 + 16 * 64 * 32 * 2;   // 100 KB
  attn_head_softmax<<<dim3(SEQ / 32), dim3(512), smem, stream>>>(Qp, Kp, Vp, ctx);

  gemm_xwt_bf16<float><<<grd, blk, 0, stream>>>(ctx, wob, bo, (float*)d_out,
                                                SEQ, DIM, DIM);
}
